// MixtureOfExperts_64570538328572
// MI455X (gfx1250) — matrix operand order repaired and checked by reading, not compiled, not run on hardware
//
#include <hip/hip_runtime.h>
#include <hip/hip_bf16.h>

#define B_TOK    32768
#define DIM      256
#define NEXP     8
#define TILE_M   128
#define NTHREADS 512
#define LDK      264   // padded bf16 row stride (elements) -> 528B, 16B aligned, bank-staggered

// Workspace layout (bf16 elements): 25 transposed weight matrices then x.
//   slot 0      : gw1^T           (256x256)
//   slots 1..8  : w1[e]^T
//   slots 9..16 : w2[e]^T
//   slots 17..24: w3[e]^T
//   offset 25*65536: x as bf16, row-major [B_TOK][256]
#define WS_MAT_ELEMS 65536
#define WS_X_OFF     (25 * WS_MAT_ELEMS)

typedef __attribute__((ext_vector_type(16))) __bf16        v16bf;
typedef __attribute__((ext_vector_type(8)))  float         v8f;
typedef __attribute__((ext_vector_type(4)))  unsigned int  u32x4;
typedef __attribute__((ext_vector_type(4)))  int           v4i;

__device__ __forceinline__ unsigned short f2bf(float f) {
    union { float f; unsigned int u; } v; v.f = f;
    unsigned int u = v.u;
    return (unsigned short)((u + 0x7FFFu + ((u >> 16) & 1u)) >> 16);  // RNE
}

// ---------------------------------------------------------------------------
// CDNA5 async global->LDS copy path (ASYNCcnt), with safe fallback.
// Builtin prototype (per hipcc diagnostic): (int4* gsrc, int4* ldst, imm, imm).
// ---------------------------------------------------------------------------
#if defined(__has_builtin)
#if __has_builtin(__builtin_amdgcn_global_load_async_to_lds_b128)
#define HAVE_ASYNC_LDS 1
#endif
#endif

__device__ __forceinline__ void cp16_to_lds(unsigned short* ldst, const unsigned short* gsrc) {
#ifdef HAVE_ASYNC_LDS
    __builtin_amdgcn_global_load_async_to_lds_b128((v4i*)gsrc, (v4i*)ldst, 0, 0);
#else
    *(u32x4*)ldst = *(const u32x4*)gsrc;
#endif
}

__device__ __forceinline__ void wait_async0() {
#ifdef HAVE_ASYNC_LDS
#if __has_builtin(__builtin_amdgcn_s_wait_asynccnt)
    __builtin_amdgcn_s_wait_asynccnt(0);
#else
    asm volatile("s_wait_asynccnt 0x0" ::: "memory");
#endif
#endif
}

// A-matrix 16x32 bf16 frag (ISA 7.12.2): lane L: M=L%16; K = kb*32 + (L/16)*8 + i (i=0..7),
// and K = kb*32 + 16 + (L/16)*8 + (i-8) (i=8..15). Two 16B LDS loads.
__device__ __forceinline__ v16bf ld_fragA(const unsigned short* row, int kb, int hlf) {
    const unsigned short* p = row + kb * 32 + hlf * 8;
    union { v16bf v; u32x4 u[2]; } f;
    f.u[0] = *(const u32x4*)(p);
    f.u[1] = *(const u32x4*)(p + 16);
    return f.v;
}
// B-matrix 32x16 bf16 frag: lane L: N=L%16; K = kb*32 + (L/16)*8 + i (i=0..7),
// and K = kb*32 + 16 + (L/16)*8 + (i-8) (i=8..15). Two 16B loads from the K-major (transposed) row.
__device__ __forceinline__ v16bf ld_fragB(const unsigned short* row, int kb, int hlf) {
    const unsigned short* p = row + kb * 32 + hlf * 8;
    union { v16bf v; u32x4 u[2]; } f;
    f.u[0] = *(const u32x4*)(p);
    f.u[1] = *(const u32x4*)(p + 16);
    return f.v;
}

#define WMMA_BF16(a, b, c) \
    __builtin_amdgcn_wmma_f32_16x16x32_bf16(false, (a), false, (b), (short)0, (c), false, false)

// 8-tile GEMM slab: acc[j] += A(mRow..+15, :) x B(:, nBase+16j..+15), K = 256.
// Next B-frag load is issued before the current WMMA to cover LDS latency.
__device__ __forceinline__ void gemm_slab(const unsigned short* sA, const unsigned short* sB,
                                          int mRow, int nBase, int r, int hlf, v8f acc[8]) {
    const unsigned short* aRow = sA + (mRow + r) * LDK;
    for (int kb = 0; kb < 8; ++kb) {
        v16bf a = ld_fragA(aRow, kb, hlf);
        v16bf b = ld_fragB(sB + (nBase + r) * LDK, kb, hlf);
        #pragma unroll
        for (int j = 0; j < 8; ++j) {
            v16bf bn = b;
            if (j < 7) bn = ld_fragB(sB + (nBase + (j + 1) * 16 + r) * LDK, kb, hlf);
            acc[j] = WMMA_BF16(a, b, acc[j]);
            b = bn;
        }
    }
}

// ---------------------------------------------------------------------------
// Prep kernel 1: transpose + convert the 25 weight matrices to bf16 in ws.
//   dst[m][n*256+k] = bf16(src[m][k*256+n])   (32x32 LDS-tiled transpose)
// ---------------------------------------------------------------------------
__global__ __launch_bounds__(256)
void prep_weights_kernel(const float* __restrict__ gw1,
                         const float* __restrict__ w1,
                         const float* __restrict__ w2,
                         const float* __restrict__ w3,
                         unsigned short* __restrict__ wsT) {
    int m = blockIdx.y;  // 0..24
    const float* src = (m == 0)  ? gw1
                     : (m <= 8)  ? (w1 + (size_t)(m - 1)  * WS_MAT_ELEMS)
                     : (m <= 16) ? (w2 + (size_t)(m - 9)  * WS_MAT_ELEMS)
                                 : (w3 + (size_t)(m - 17) * WS_MAT_ELEMS);
    unsigned short* dst = wsT + (size_t)m * WS_MAT_ELEMS;

    __shared__ float tile[32][33];
    int bx = blockIdx.x & 7, by = blockIdx.x >> 3;      // 8x8 tiles of 32x32
    int tx = threadIdx.x & 31, ty = threadIdx.x >> 5;   // 32x8 threads
    #pragma unroll
    for (int yy = 0; yy < 32; yy += 8)
        tile[ty + yy][tx] = src[(size_t)(by * 32 + ty + yy) * 256 + bx * 32 + tx];
    __syncthreads();
    #pragma unroll
    for (int yy = 0; yy < 32; yy += 8)
        dst[(size_t)(bx * 32 + ty + yy) * 256 + by * 32 + tx] = f2bf(tile[tx][ty + yy]);
}

// Prep kernel 2: x f32 -> bf16 (same layout).
__global__ __launch_bounds__(256)
void prep_x_kernel(const float* __restrict__ x, unsigned short* __restrict__ xbf) {
    size_t i = ((size_t)blockIdx.x * 256 + threadIdx.x) * 8;
    #pragma unroll
    for (int q = 0; q < 8; ++q) xbf[i + q] = f2bf(x[i + q]);
}

// ---------------------------------------------------------------------------
// Fused MoE kernel: 1 workgroup = 128 tokens, 16 waves, bf16 WMMA throughout.
// ---------------------------------------------------------------------------
__global__ __launch_bounds__(NTHREADS)
void moe_fused_kernel(const unsigned short* __restrict__ wsT,   // prepped weights + x (bf16)
                      const float* __restrict__ gb1,
                      const float* __restrict__ gw2, const float* __restrict__ gb2,
                      const float* __restrict__ b1,  const float* __restrict__ g1,
                      const float* __restrict__ be1,
                      const float* __restrict__ b2,  const float* __restrict__ g2,
                      const float* __restrict__ be2,
                      const float* __restrict__ b3,
                      float* __restrict__ out)
{
    __shared__ __align__(16) unsigned short sXA[TILE_M * LDK];  //  66 KB: x tile, bf16 A-layout
    __shared__ __align__(16) unsigned short sHA[TILE_M * LDK];  //  66 KB: hidden, bf16 A-layout
    __shared__ __align__(16) unsigned short sWB[DIM * LDK];     // 132 KB: weight^T bf16 (K-major)
    __shared__ float sP[3 * DIM];          // bias / gamma / beta of current layer
    __shared__ float sGW2[DIM * NEXP];
    __shared__ float sB3[NEXP * DIM];
    __shared__ float sGB2[NEXP];
    __shared__ float sLog[TILE_M * NEXP];
    __shared__ float sGate[TILE_M * NEXP]; // normalized top-2 weights, 0 elsewhere
    __shared__ float sRS[TILE_M * 2];
    __shared__ float sRQ[TILE_M * 2];

    const int tid   = threadIdx.x;
    const int lane  = tid & 31;
    const int wave  = tid >> 5;
    const int r     = lane & 15;
    const int hlf   = lane >> 4;
    const int mRow  = (wave & 7) * 16;
    const int nHalf = wave >> 3;
    const int nBase = nHalf * 128;
    const int tok0  = blockIdx.x * TILE_M;

    const unsigned short* gw1T = wsT;
    const unsigned short* w1T  = wsT + (size_t)1  * WS_MAT_ELEMS;
    const unsigned short* w2T  = wsT + (size_t)9  * WS_MAT_ELEMS;
    const unsigned short* w3T  = wsT + (size_t)17 * WS_MAT_ELEMS;
    const unsigned short* xbf  = wsT + WS_X_OFF + (size_t)tok0 * DIM;

    // ---------------- stage x tile (pure bf16 copies), small params, gw1^T ----------------
    #pragma unroll
    for (int it = 0; it < 8; ++it) {                    // 4096 16B chunks of x tile
        int c = tid + it * NTHREADS;
        int t = c >> 5, d = (c & 31) << 3;
        cp16_to_lds(sXA + t * LDK + d, xbf + t * 256 + d);
    }
    #pragma unroll
    for (int it = 0; it < 16; ++it) {                   // 8192 16B chunks of gw1^T
        int c = tid + it * NTHREADS;
        int n = c >> 5, k = (c & 31) << 3;
        cp16_to_lds(sWB + n * LDK + k, gw1T + n * 256 + k);
    }
    for (int i = tid; i < DIM * NEXP; i += NTHREADS) sGW2[i] = gw2[i];
    for (int i = tid; i < NEXP * DIM; i += NTHREADS) sB3[i] = b3[i];
    if (tid < NEXP) sGB2[tid] = gb2[tid];
    for (int i = tid; i < DIM; i += NTHREADS) sP[i] = gb1[i];
    wait_async0();
    __syncthreads();

    v8f acc[8];

    // ---------------- gating GEMM: relu(x@gw1 + gb1) -> sHA ----------------
    #pragma unroll
    for (int j = 0; j < 8; ++j)
        #pragma unroll
        for (int v = 0; v < 8; ++v) acc[j][v] = 0.0f;
    gemm_slab(sXA, sWB, mRow, nBase, r, hlf, acc);
    #pragma unroll
    for (int j = 0; j < 8; ++j) {
        int col = nBase + j * 16 + r;
        float bias = sP[col];
        #pragma unroll
        for (int v = 0; v < 8; ++v) {
            float val = acc[j][v] + bias;
            val = val > 0.f ? val : 0.f;
            sHA[(mRow + v + hlf * 8) * LDK + col] = f2bf(val);
        }
    }
    __syncthreads();

    // ---------------- logits = hg @ gw2 + gb2 (vectorized bf16 reads) ----------------
    for (int task = tid; task < TILE_M * NEXP; task += NTHREADS) {
        int t = task >> 3, e = task & 7;
        const unsigned short* hrow = sHA + t * LDK;
        float s = 0.f;
        for (int kc = 0; kc < DIM; kc += 8) {
            u32x4 h = *(const u32x4*)(hrow + kc);
            #pragma unroll
            for (int q = 0; q < 4; ++q) {
                unsigned int u = h[q];
                float lo = __uint_as_float(u << 16);
                float hi = __uint_as_float(u & 0xffff0000u);
                s += lo * sGW2[(kc + 2 * q) * NEXP + e] + hi * sGW2[(kc + 2 * q + 1) * NEXP + e];
            }
        }
        sLog[task] = s + sGB2[e];
    }
    __syncthreads();

    // ---- softmax + top-2 + renorm (tid<128); concurrently stage w1[0]^T + params1[0]
    if (tid < TILE_M) {
        float l[8];
        #pragma unroll
        for (int e = 0; e < 8; ++e) l[e] = sLog[tid * 8 + e];
        float mx = l[0];
        #pragma unroll
        for (int e = 1; e < 8; ++e) mx = fmaxf(mx, l[e]);
        float g[8];
        #pragma unroll
        for (int e = 0; e < 8; ++e) g[e] = __expf(l[e] - mx);
        int i0 = 0;
        #pragma unroll
        for (int e = 1; e < 8; ++e) if (g[e] > g[i0]) i0 = e;
        int i1 = (i0 == 0) ? 1 : 0;
        #pragma unroll
        for (int e = 0; e < 8; ++e) if (e != i0 && g[e] > g[i1]) i1 = e;
        float s2 = g[i0] + g[i1];
        #pragma unroll
        for (int e = 0; e < 8; ++e)
            sGate[tid * 8 + e] = (e == i0) ? g[i0] / s2 : ((e == i1) ? g[i1] / s2 : 0.f);
    }
    #pragma unroll
    for (int it = 0; it < 16; ++it) {
        int c = tid + it * NTHREADS;
        int n = c >> 5, k = (c & 31) << 3;
        cp16_to_lds(sWB + n * LDK + k, w1T + n * 256 + k);
    }
    for (int i = tid; i < DIM; i += NTHREADS) {
        sP[i] = b1[i]; sP[DIM + i] = g1[i]; sP[2 * DIM + i] = be1[i];
    }
    wait_async0();
    __syncthreads();

    // ---------------- expert loop: out += gate_e * MLP_e(x) ----------------
    v8f outAcc[8];
    #pragma unroll
    for (int j = 0; j < 8; ++j)
        #pragma unroll
        for (int v = 0; v < 8; ++v) outAcc[j][v] = 0.0f;

    for (int e = 0; e < NEXP; ++e) {
        // ===== layer 1: relu(LN(x@w1 + b1)) =====
        #pragma unroll
        for (int j = 0; j < 8; ++j)
            #pragma unroll
            for (int v = 0; v < 8; ++v) acc[j][v] = 0.0f;
        gemm_slab(sXA, sWB, mRow, nBase, r, hlf, acc);

        float gv[8], bv[8];
        #pragma unroll
        for (int j = 0; j < 8; ++j) {
            int col = nBase + j * 16 + r;
            float bias = sP[col];
            gv[j] = sP[DIM + col]; bv[j] = sP[2 * DIM + col];
            #pragma unroll
            for (int v = 0; v < 8; ++v) acc[j][v] += bias;
        }
        #pragma unroll
        for (int v = 0; v < 8; ++v) {          // per-row partial stats over this wave's 128 cols
            float s = 0.f, q = 0.f;
            #pragma unroll
            for (int j = 0; j < 8; ++j) { float a_ = acc[j][v]; s += a_; q += a_ * a_; }
            #pragma unroll
            for (int m = 1; m < 16; m <<= 1) { s += __shfl_xor(s, m); q += __shfl_xor(q, m); }
            if (r == 0) {
                int row = mRow + v + hlf * 8;
                sRS[row * 2 + nHalf] = s;
                sRQ[row * 2 + nHalf] = q;
            }
        }
        __syncthreads();
        #pragma unroll
        for (int v = 0; v < 8; ++v) {
            int row = mRow + v + hlf * 8;
            float mean = (sRS[row * 2] + sRS[row * 2 + 1]) * (1.f / 256.f);
            float var  = (sRQ[row * 2] + sRQ[row * 2 + 1]) * (1.f / 256.f) - mean * mean;
            float rstd = rsqrtf(var + 1e-5f);
            #pragma unroll
            for (int j = 0; j < 8; ++j) {
                int col = nBase + j * 16 + r;
                float val = (acc[j][v] - mean) * rstd * gv[j] + bv[j];
                val = val > 0.f ? val : 0.f;
                sHA[row * LDK + col] = f2bf(val);
            }
        }
        #pragma unroll
        for (int it = 0; it < 16; ++it) {      // stage w2[e]^T
            int c = tid + it * NTHREADS;
            int n = c >> 5, k = (c & 31) << 3;
            cp16_to_lds(sWB + n * LDK + k, w2T + (size_t)e * WS_MAT_ELEMS + n * 256 + k);
        }
        for (int i = tid; i < DIM; i += NTHREADS) {
            sP[i] = b2[e * DIM + i]; sP[DIM + i] = g2[e * DIM + i]; sP[2 * DIM + i] = be2[e * DIM + i];
        }
        wait_async0();
        __syncthreads();

        // ===== layer 2: gate_e * relu(LN(h1@w2 + b2)) =====
        #pragma unroll
        for (int j = 0; j < 8; ++j)
            #pragma unroll
            for (int v = 0; v < 8; ++v) acc[j][v] = 0.0f;
        gemm_slab(sHA, sWB, mRow, nBase, r, hlf, acc);

        float rsc[8];
        #pragma unroll
        for (int j = 0; j < 8; ++j) {
            int col = nBase + j * 16 + r;
            float bias = sP[col];
            gv[j] = sP[DIM + col]; bv[j] = sP[2 * DIM + col];
            #pragma unroll
            for (int v = 0; v < 8; ++v) acc[j][v] += bias;
        }
        #pragma unroll
        for (int v = 0; v < 8; ++v) {
            rsc[v] = sGate[(mRow + v + hlf * 8) * 8 + e];
            float s = 0.f, q = 0.f;
            #pragma unroll
            for (int j = 0; j < 8; ++j) { float a_ = acc[j][v]; s += a_; q += a_ * a_; }
            #pragma unroll
            for (int m = 1; m < 16; m <<= 1) { s += __shfl_xor(s, m); q += __shfl_xor(q, m); }
            if (r == 0) {
                int row = mRow + v + hlf * 8;
                sRS[row * 2 + nHalf] = s;
                sRQ[row * 2 + nHalf] = q;
            }
        }
        __syncthreads();
        #pragma unroll
        for (int v = 0; v < 8; ++v) {
            int row = mRow + v + hlf * 8;
            float mean = (sRS[row * 2] + sRS[row * 2 + 1]) * (1.f / 256.f);
            float var  = (sRQ[row * 2] + sRQ[row * 2 + 1]) * (1.f / 256.f) - mean * mean;
            float rstd = rsqrtf(var + 1e-5f);
            #pragma unroll
            for (int j = 0; j < 8; ++j) {
                int col = nBase + j * 16 + r;
                float val = (acc[j][v] - mean) * rstd * gv[j] + bv[j];
                val = val > 0.f ? val : 0.f;
                sHA[row * LDK + col] = f2bf(val * rsc[v]);  // pre-scale rows by gate weight
            }
        }
        #pragma unroll
        for (int it = 0; it < 16; ++it) {      // stage w3[e]^T
            int c = tid + it * NTHREADS;
            int n = c >> 5, k = (c & 31) << 3;
            cp16_to_lds(sWB + n * LDK + k, w3T + (size_t)e * WS_MAT_ELEMS + n * 256 + k);
        }
        wait_async0();
        __syncthreads();

        // ===== layer 3: outAcc += (gate_e*h2) @ w3 =====
        gemm_slab(sHA, sWB, mRow, nBase, r, hlf, outAcc);
        __syncthreads();
        if (e < NEXP - 1) {                    // stage w1[e+1]^T + params1
            #pragma unroll
            for (int it = 0; it < 16; ++it) {
                int c = tid + it * NTHREADS;
                int n = c >> 5, k = (c & 31) << 3;
                cp16_to_lds(sWB + n * LDK + k, w1T + (size_t)(e + 1) * WS_MAT_ELEMS + n * 256 + k);
            }
            for (int i = tid; i < DIM; i += NTHREADS) {
                sP[i] = b1[(e + 1) * DIM + i];
                sP[DIM + i] = g1[(e + 1) * DIM + i];
                sP[2 * DIM + i] = be1[(e + 1) * DIM + i];
            }
        }
        wait_async0();
        __syncthreads();
    }

    // ---------------- epilogue: + sum_e gate_e * b3[e] ----------------
    #pragma unroll
    for (int v = 0; v < 8; ++v) {
        int row = mRow + v + hlf * 8;
        #pragma unroll
        for (int j = 0; j < 8; ++j) {
            int col = nBase + j * 16 + r;
            float bias = 0.f;
            #pragma unroll
            for (int e = 0; e < 8; ++e) bias += sGate[row * 8 + e] * sB3[e * DIM + col];
            out[(size_t)(tok0 + row) * DIM + col] = outAcc[j][v] + bias;
        }
    }
}

extern "C" void kernel_launch(void* const* d_in, const int* in_sizes, int n_in,
                              void* d_out, int out_size, void* d_ws, size_t ws_size,
                              hipStream_t stream) {
    (void)in_sizes; (void)n_in; (void)ws_size; (void)out_size;
    const float* x   = (const float*)d_in[0];
    const float* gw1 = (const float*)d_in[1];
    const float* gb1 = (const float*)d_in[2];
    const float* gw2 = (const float*)d_in[3];
    const float* gb2 = (const float*)d_in[4];
    const float* w1  = (const float*)d_in[5];
    const float* b1  = (const float*)d_in[6];
    const float* g1  = (const float*)d_in[7];
    const float* be1 = (const float*)d_in[8];
    const float* w2  = (const float*)d_in[9];
    const float* b2  = (const float*)d_in[10];
    const float* g2  = (const float*)d_in[11];
    const float* be2 = (const float*)d_in[12];
    const float* w3  = (const float*)d_in[13];
    const float* b3  = (const float*)d_in[14];
    float* out = (float*)d_out;
    unsigned short* wsT = (unsigned short*)d_ws;   // needs 25*64K + 32768*256 bf16 = ~20 MB

    // Prep: transpose+convert weights, convert x (runs every call; deterministic).
    prep_weights_kernel<<<dim3(64, 25), 256, 0, stream>>>(gw1, w1, w2, w3, wsT);
    prep_x_kernel<<<dim3((B_TOK * DIM) / (256 * 8)), 256, 0, stream>>>(x, wsT + WS_X_OFF);

    // Fused MoE: 256 workgroups, one 128-token tile each.
    moe_fused_kernel<<<dim3(B_TOK / TILE_M), NTHREADS, 0, stream>>>(
        wsT, gb1, gw2, gb2, b1, g1, be1, b2, g2, be2, b3, out);
}
